// CrossAttentionBlock_37022618092055
// MI455X (gfx1250) — matrix operand order repaired and checked by reading, not compiled, not run on hardware
//
#include <hip/hip_runtime.h>

typedef __attribute__((ext_vector_type(16))) __bf16 v16bf;
typedef __attribute__((ext_vector_type(8)))  __bf16 v8bf;
typedef __attribute__((ext_vector_type(8)))  float  v8f;

#define B_  2
#define N_  256
#define DC_ 96
#define DP_ 160
#define DE_ 64
#define H_  8

// ---------------------------------------------------------------------------
// WMMA fragment loaders (wave32 layouts per CDNA5 ISA 7.12.2)
// ---------------------------------------------------------------------------
// A (16x32 bf16, row-major source, lda elements, lda % 8 == 0):
//   lane L: row = mbase + (L&15); elems 0..7 -> K = kstep + (L>>4)*8 + i
//                                 elems 8..15 -> K = kstep + 16 + (L>>4)*8 + i
__device__ __forceinline__ v16bf load_a_frag(const __bf16* A, int lda,
                                             int mbase, int kstep, int lane) {
  int row = mbase + (lane & 15);
  int kg  = (lane >> 4) * 8;
  const __bf16* p = A + row * lda + kstep + kg;
  union { v16bf v; v8bf h[2]; } u;
  u.h[0] = *(const v8bf*)(p);
  u.h[1] = *(const v8bf*)(p + 16);
  return u.v;
}

// B (32x16 bf16) from W^T stored [N][K] (ldb % 8 == 0):
//   lane L: col = nbase + (L&15); elems 0..7 -> K = kstep + (L>>4)*8 + i
//                                 elems 8..15 -> K = kstep + 16 + (L>>4)*8 + i
__device__ __forceinline__ v16bf load_b_frag(const __bf16* BT, int ldb,
                                             int nbase, int kstep, int lane) {
  int col = nbase + (lane & 15);
  int kb  = kstep + (lane >> 4) * 8;
  const __bf16* p = BT + col * ldb + kb;
  union { v16bf v; v8bf h[2]; } u;
  u.h[0] = *(const v8bf*)(p);
  u.h[1] = *(const v8bf*)(p + 16);
  return u.v;
}

__device__ __forceinline__ v8f gemm_tile(const __bf16* A, int lda,
                                         const __bf16* BT, int ldb, int K,
                                         int mbase, int nbase, v8f acc, int lane) {
  for (int k = 0; k < K; k += 32) {
    v16bf a = load_a_frag(A, lda, mbase, k, lane);
    v16bf b = load_b_frag(BT, ldb, nbase, k, lane);
    acc = __builtin_amdgcn_wmma_f32_16x16x32_bf16(false, a, false, b,
                                                  (short)0, acc, false, false);
  }
  return acc;
}

// ---------------------------------------------------------------------------
// Kernel 0: f32 [K][N] -> bf16 transposed [N][K]
// ---------------------------------------------------------------------------
__global__ void k_wT(const float* __restrict__ src, __bf16* __restrict__ dst,
                     int K, int Nc) {
  int idx = blockIdx.x * blockDim.x + threadIdx.x;
  if (idx >= K * Nc) return;
  int n = idx / K;
  int k = idx - n * K;
  dst[n * K + k] = (__bf16)src[k * Nc + n];
}

// ---------------------------------------------------------------------------
// Kernel 1: QKV projections (f32, tiny)
// qkvc: [B*N][288] = classes @ wc_qkv + bc ; qkvp: [B*N][480]
// ---------------------------------------------------------------------------
__global__ void k_qkv(const float* __restrict__ classes, const float* __restrict__ params,
                      const float* __restrict__ wc, const float* __restrict__ bc,
                      const float* __restrict__ wp, const float* __restrict__ bp,
                      float* __restrict__ qkvc, float* __restrict__ qkvp) {
  int idx = blockIdx.x * blockDim.x + threadIdx.x;
  const int TOTC = B_ * N_ * 3 * DC_;
  if (idx < TOTC) {
    int row = idx / (3 * DC_), col = idx - row * (3 * DC_);
    float acc = bc[col];
    for (int k = 0; k < DC_; ++k)
      acc += classes[row * DC_ + k] * wc[k * (3 * DC_) + col];
    qkvc[idx] = acc;
  } else {
    idx -= TOTC;
    if (idx < B_ * N_ * 3 * DP_) {
      int row = idx / (3 * DP_), col = idx - row * (3 * DP_);
      float acc = bp[col];
      for (int k = 0; k < DP_; ++k)
        acc += params[row * DP_ + k] * wp[k * (3 * DP_) + col];
      qkvp[idx] = acc;
    }
  }
}

// ---------------------------------------------------------------------------
// Kernel 2: fused FiLM MLPs + concat edge projection + score sums
// One block = 8 waves handles (b, q, 32-row k-tile)
// ---------------------------------------------------------------------------
__global__ __launch_bounds__(256)
void k_main(const float* __restrict__ edges,
            const float* __restrict__ qkvc, const float* __restrict__ qkvp,
            const __bf16* __restrict__ fcmulT, const __bf16* __restrict__ fcaddT,
            const __bf16* __restrict__ fcm1T,  const __bf16* __restrict__ fcm2T,
            const __bf16* __restrict__ fpmulT, const __bf16* __restrict__ fpaddT,
            const __bf16* __restrict__ fpm1T,  const __bf16* __restrict__ fpm2T,
            const __bf16* __restrict__ weT,
            const float* __restrict__ fc_mul_b, const float* __restrict__ fc_add_b,
            const float* __restrict__ fc_m1_b,  const float* __restrict__ fc_m2_b,
            const float* __restrict__ fp_mul_b, const float* __restrict__ fp_add_b,
            const float* __restrict__ fp_m1_b,  const float* __restrict__ fp_m2_b,
            const float* __restrict__ be_out,
            float* __restrict__ scoreC, float* __restrict__ scoreP,
            float* __restrict__ out_e) {
  __shared__ __bf16 sE[32 * 72];    // edge tile (stride 72)
  __shared__ __bf16 sX[32 * 168];   // activations (stride 168)
  __shared__ __bf16 sY[32 * 168];
  __shared__ __bf16 sC[32 * 264];   // interleaved concat (stride 264)

  const int tid  = threadIdx.x;
  const int lane = tid & 31;
  const int wave = tid >> 5;
  const int bid  = blockIdx.x;
  const int kt = bid & 7;
  const int qi = (bid >> 3) & (N_ - 1);
  const int bb = bid >> 11;
  const int k0 = kt * 32;

  const float inv_dc = 0.10206207261596575f;  // 1/sqrt(96)
  const float inv_dp = 0.07905694150420949f;  // 1/sqrt(160)

  // ---- load edge tile (contiguous 32*64 floats) -> bf16 LDS
  const float* ePtr = edges + (((size_t)(bb * N_ + qi)) * N_ + k0) * DE_;
  for (int i = tid; i < 32 * DE_; i += 256) {
    int row = i >> 6, col = i & 63;
    sE[row * 72 + col] = (__bf16)ePtr[i];
  }
  __syncthreads();

  const float* qcRow = qkvc + (size_t)(bb * N_ + qi) * (3 * DC_);
  const float* qpRow = qkvp + (size_t)(bb * N_ + qi) * (3 * DP_);

  // ========================== CLASS path (DC=96) ==========================
  // FiLM combine: x = a*mul + add + a, a = Qc*Kc/sqrt(DC)
  for (int t = wave; t < 12; t += 8) {
    int mi = t & 1, ni = t >> 1;
    int col = ni * 16 + (lane & 15);
    v8f am, aa;
    float bm = fc_mul_b[col], ba = fc_add_b[col];
    for (int r = 0; r < 8; ++r) { am[r] = bm; aa[r] = ba; }
    am = gemm_tile(sE, 72, fcmulT, 64, 64, mi * 16, ni * 16, am, lane);
    aa = gemm_tile(sE, 72, fcaddT, 64, 64, mi * 16, ni * 16, aa, lane);
    float qv = qcRow[col] * inv_dc;
    int rbase = mi * 16 + ((lane >> 4) << 3);
    for (int r = 0; r < 8; ++r) {
      int krow = k0 + rbase + r;
      float kc = qkvc[((size_t)(bb * N_ + krow)) * (3 * DC_) + DC_ + col];
      float a = qv * kc;
      float x = a * am[r] + aa[r] + a;
      sX[(rbase + r) * 168 + col] = (__bf16)x;
    }
  }
  __syncthreads();

  for (int t = wave; t < 12; t += 8) {        // h1 = lrelu(x @ m1 + b1)
    int mi = t & 1, ni = t >> 1;
    int col = ni * 16 + (lane & 15);
    v8f acc; float bv = fc_m1_b[col];
    for (int r = 0; r < 8; ++r) acc[r] = bv;
    acc = gemm_tile(sX, 168, fcm1T, 96, 96, mi * 16, ni * 16, acc, lane);
    int rbase = mi * 16 + ((lane >> 4) << 3);
    for (int r = 0; r < 8; ++r) {
      float x = acc[r]; x = x > 0.0f ? x : 0.05f * x;
      sY[(rbase + r) * 168 + col] = (__bf16)x;
    }
  }
  __syncthreads();

  for (int t = wave; t < 12; t += 8) {        // attn_class = lrelu(h1 @ m2 + b2)
    int mi = t & 1, ni = t >> 1;
    int col = ni * 16 + (lane & 15);
    v8f acc; float bv = fc_m2_b[col];
    for (int r = 0; r < 8; ++r) acc[r] = bv;
    acc = gemm_tile(sY, 168, fcm2T, 96, 96, mi * 16, ni * 16, acc, lane);
    int h = col / 12, d = col - h * 12;
    int cc = h * 32 + d;                      // interleaved concat column
    int rbase = mi * 16 + ((lane >> 4) << 3);
    for (int r = 0; r < 8; ++r) {
      float x = acc[r]; x = x > 0.0f ? x : 0.05f * x;
      sC[(rbase + r) * 264 + cc] = (__bf16)x;
    }
  }
  __syncthreads();

  // ========================== PARAM path (DP=160) =========================
  for (int t = wave; t < 20; t += 8) {
    int mi = t & 1, ni = t >> 1;
    int col = ni * 16 + (lane & 15);
    v8f am, aa;
    float bm = fp_mul_b[col], ba = fp_add_b[col];
    for (int r = 0; r < 8; ++r) { am[r] = bm; aa[r] = ba; }
    am = gemm_tile(sE, 72, fpmulT, 64, 64, mi * 16, ni * 16, am, lane);
    aa = gemm_tile(sE, 72, fpaddT, 64, 64, mi * 16, ni * 16, aa, lane);
    float qv = qpRow[col] * inv_dp;
    int rbase = mi * 16 + ((lane >> 4) << 3);
    for (int r = 0; r < 8; ++r) {
      int krow = k0 + rbase + r;
      float kp = qkvp[((size_t)(bb * N_ + krow)) * (3 * DP_) + DP_ + col];
      float a = qv * kp;
      float x = a * am[r] + aa[r] + a;
      sX[(rbase + r) * 168 + col] = (__bf16)x;
    }
  }
  __syncthreads();

  for (int t = wave; t < 20; t += 8) {
    int mi = t & 1, ni = t >> 1;
    int col = ni * 16 + (lane & 15);
    v8f acc; float bv = fp_m1_b[col];
    for (int r = 0; r < 8; ++r) acc[r] = bv;
    acc = gemm_tile(sX, 168, fpm1T, 160, 160, mi * 16, ni * 16, acc, lane);
    int rbase = mi * 16 + ((lane >> 4) << 3);
    for (int r = 0; r < 8; ++r) {
      float x = acc[r]; x = x > 0.0f ? x : 0.05f * x;
      sY[(rbase + r) * 168 + col] = (__bf16)x;
    }
  }
  __syncthreads();

  for (int t = wave; t < 20; t += 8) {
    int mi = t & 1, ni = t >> 1;
    int col = ni * 16 + (lane & 15);
    v8f acc; float bv = fp_m2_b[col];
    for (int r = 0; r < 8; ++r) acc[r] = bv;
    acc = gemm_tile(sY, 168, fpm2T, 160, 160, mi * 16, ni * 16, acc, lane);
    int h = col / 20, d = col - h * 20;
    int cc = h * 32 + 12 + d;
    int rbase = mi * 16 + ((lane >> 4) << 3);
    for (int r = 0; r < 8; ++r) {
      float x = acc[r]; x = x > 0.0f ? x : 0.05f * x;
      sC[(rbase + r) * 264 + cc] = (__bf16)x;
    }
  }
  __syncthreads();

  // ---- per-(row, head) score sums (32 rows x 8 heads = 256 threads)
  {
    int row = tid >> 3, h = tid & 7;
    const __bf16* rp = sC + row * 264 + h * 32;
    float s1 = 0.0f, s2 = 0.0f;
    for (int d = 0; d < 12; ++d) s1 += (float)rp[d];
    for (int d = 0; d < 20; ++d) s2 += (float)rp[12 + d];
    size_t so = (((size_t)(bb * N_ + qi)) * N_ + (k0 + row)) * H_ + h;
    scoreC[so] = s1;
    scoreP[so] = s2;
  }

  // ---- out_e = concat(32x256) @ we_out(256x64) + be_out
  for (int t = wave; t < 8; t += 8) {
    int mi = t & 1, ni = t >> 1;
    int col = ni * 16 + (lane & 15);
    v8f acc; float bv = be_out[col];
    for (int r = 0; r < 8; ++r) acc[r] = bv;
    acc = gemm_tile(sC, 264, weT, 256, 256, mi * 16, ni * 16, acc, lane);
    int rbase = mi * 16 + ((lane >> 4) << 3);
    for (int r = 0; r < 8; ++r) {
      size_t o = (((size_t)(bb * N_ + qi)) * N_ + (k0 + rbase + r)) * DE_ + col;
      out_e[o] = acc[r];
    }
  }
}

// ---------------------------------------------------------------------------
// Kernel 3: softmax over k + cross-weighted V + output projections
// One block per (b, q)
// ---------------------------------------------------------------------------
__global__ __launch_bounds__(256)
void k_attn_out(const float* __restrict__ scoreC, const float* __restrict__ scoreP,
                const float* __restrict__ qkvc, const float* __restrict__ qkvp,
                const float* __restrict__ wc_out, const float* __restrict__ bc_out,
                const float* __restrict__ wp_out, const float* __restrict__ bp_out,
                float* __restrict__ out_c, float* __restrict__ out_p) {
  __shared__ float shC[H_ * N_];
  __shared__ float shP[H_ * N_];
  __shared__ float mC[H_], sCd[H_], mP[H_], sPd[H_];
  __shared__ float wcs[DC_], wps[DP_];

  int tid = threadIdx.x;
  int bq = blockIdx.x;            // b*N + q
  int bb = bq >> 8;               // N_ == 256
  size_t base = (size_t)bq * N_ * H_;

  for (int i = tid; i < N_ * H_; i += 256) {
    int k = i >> 3, h = i & 7;
    shC[h * N_ + k] = scoreC[base + (size_t)k * H_ + h];
    shP[h * N_ + k] = scoreP[base + (size_t)k * H_ + h];
  }
  __syncthreads();

  {
    int h = tid >> 5, ln = tid & 31;  // 8 waves -> 8 heads
    float m = -3.0e38f;
    for (int j = 0; j < 8; ++j) m = fmaxf(m, shC[h * N_ + ln + 32 * j]);
    for (int off = 16; off; off >>= 1) m = fmaxf(m, __shfl_xor(m, off, 32));
    float s = 0.0f;
    for (int j = 0; j < 8; ++j) s += __expf(shC[h * N_ + ln + 32 * j] - m);
    for (int off = 16; off; off >>= 1) s += __shfl_xor(s, off, 32);
    if (ln == 0) { mC[h] = m; sCd[h] = s; }

    m = -3.0e38f;
    for (int j = 0; j < 8; ++j) m = fmaxf(m, shP[h * N_ + ln + 32 * j]);
    for (int off = 16; off; off >>= 1) m = fmaxf(m, __shfl_xor(m, off, 32));
    s = 0.0f;
    for (int j = 0; j < 8; ++j) s += __expf(shP[h * N_ + ln + 32 * j] - m);
    for (int off = 16; off; off >>= 1) s += __shfl_xor(s, off, 32);
    if (ln == 0) { mP[h] = m; sPd[h] = s; }
  }
  __syncthreads();

  // weighted_classes uses softmax(param scores) with Vc; weighted_params uses
  // softmax(class scores) with Vp (cross, per reference).
  if (tid < DC_) {
    int c = tid, h = c / 12;
    float m = mP[h], inv = 1.0f / sPd[h];
    float acc = 0.0f;
    const float* v = qkvc + (size_t)bb * N_ * (3 * DC_) + 2 * DC_ + c;
    for (int k = 0; k < N_; ++k)
      acc += __expf(shP[h * N_ + k] - m) * inv * v[(size_t)k * (3 * DC_)];
    wcs[c] = acc;
  } else {
    int c = tid - DC_, h = c / 20;
    float m = mC[h], inv = 1.0f / sCd[h];
    float acc = 0.0f;
    const float* v = qkvp + (size_t)bb * N_ * (3 * DP_) + 2 * DP_ + c;
    for (int k = 0; k < N_; ++k)
      acc += __expf(shC[h * N_ + k] - m) * inv * v[(size_t)k * (3 * DP_)];
    wps[c] = acc;
  }
  __syncthreads();

  if (tid < DC_) {
    float acc = bc_out[tid];
    for (int j = 0; j < DC_; ++j) acc += wcs[j] * wc_out[j * DC_ + tid];
    out_c[(size_t)bq * DC_ + tid] = acc;
  } else {
    int c = tid - DC_;
    float acc = bp_out[c];
    for (int j = 0; j < DP_; ++j) acc += wps[j] * wp_out[j * DP_ + c];
    out_p[(size_t)bq * DP_ + c] = acc;
  }
}

// ---------------------------------------------------------------------------
extern "C" void kernel_launch(void* const* d_in, const int* in_sizes, int n_in,
                              void* d_out, int out_size, void* d_ws, size_t ws_size,
                              hipStream_t stream) {
  const float* classes  = (const float*)d_in[0];
  const float* params   = (const float*)d_in[1];
  const float* edges    = (const float*)d_in[2];
  const float* wc_qkv   = (const float*)d_in[3];
  const float* bc_qkv   = (const float*)d_in[4];
  const float* wp_qkv   = (const float*)d_in[5];
  const float* bp_qkv   = (const float*)d_in[6];
  const float* fc_mul_w = (const float*)d_in[7];
  const float* fc_mul_b = (const float*)d_in[8];
  const float* fc_add_w = (const float*)d_in[9];
  const float* fc_add_b = (const float*)d_in[10];
  const float* fc_m1_w  = (const float*)d_in[11];
  const float* fc_m1_b  = (const float*)d_in[12];
  const float* fc_m2_w  = (const float*)d_in[13];
  const float* fc_m2_b  = (const float*)d_in[14];
  const float* fp_mul_w = (const float*)d_in[15];
  const float* fp_mul_b = (const float*)d_in[16];
  const float* fp_add_w = (const float*)d_in[17];
  const float* fp_add_b = (const float*)d_in[18];
  const float* fp_m1_w  = (const float*)d_in[19];
  const float* fp_m1_b  = (const float*)d_in[20];
  const float* fp_m2_w  = (const float*)d_in[21];
  const float* fp_m2_b  = (const float*)d_in[22];
  const float* wc_out   = (const float*)d_in[23];
  const float* bc_out   = (const float*)d_in[24];
  const float* wp_out   = (const float*)d_in[25];
  const float* bp_out   = (const float*)d_in[26];
  const float* we_out   = (const float*)d_in[27];
  const float* be_out   = (const float*)d_in[28];
  (void)in_sizes; (void)n_in; (void)out_size; (void)ws_size;

  // ---- workspace layout (bytes)
  char* ws = (char*)d_ws;
  float* qkvc   = (float*)(ws + 0);          // 2*256*288*4 = 589824
  float* qkvp   = (float*)(ws + 589824);     // 2*256*480*4 = 983040
  float* scoreC = (float*)(ws + 1572864);    // 2*256*256*8*4 = 4194304
  float* scoreP = (float*)(ws + 5767168);    // 4194304
  __bf16* wb    = (__bf16*)(ws + 9961472);
  __bf16* fcmulT = wb;                       //  96* 64
  __bf16* fcaddT = fcmulT + 96 * 64;
  __bf16* fcm1T  = fcaddT + 96 * 64;         //  96* 96
  __bf16* fcm2T  = fcm1T + 96 * 96;
  __bf16* fpmulT = fcm2T + 96 * 96;          // 160* 64
  __bf16* fpaddT = fpmulT + 160 * 64;
  __bf16* fpm1T  = fpaddT + 160 * 64;        // 160*160
  __bf16* fpm2T  = fpm1T + 160 * 160;
  __bf16* weT    = fpm2T + 160 * 160;        //  64*256

  // ---- output layout
  float* out_c = (float*)d_out;                       // 2*256*96
  float* out_p = out_c + (size_t)B_ * N_ * DC_;       // 2*256*160
  float* out_e = out_p + (size_t)B_ * N_ * DP_;       // 2*256*256*64

  // Kernel 0: weight transpose + bf16 convert (9 small launches)
  auto wt = [&](const float* s, __bf16* d, int K, int Nc) {
    int tot = K * Nc;
    k_wT<<<(tot + 255) / 256, 256, 0, stream>>>(s, d, K, Nc);
  };
  wt(fc_mul_w, fcmulT, 64, 96);
  wt(fc_add_w, fcaddT, 64, 96);
  wt(fc_m1_w,  fcm1T,  96, 96);
  wt(fc_m2_w,  fcm2T,  96, 96);
  wt(fp_mul_w, fpmulT, 64, 160);
  wt(fp_add_w, fpaddT, 64, 160);
  wt(fp_m1_w,  fpm1T,  160, 160);
  wt(fp_m2_w,  fpm2T,  160, 160);
  wt(we_out,   weT,    256, 64);

  // Kernel 1: QKV projections
  {
    int tot = B_ * N_ * 3 * DC_ + B_ * N_ * 3 * DP_;  // 393216
    k_qkv<<<(tot + 255) / 256, 256, 0, stream>>>(classes, params,
                                                 wc_qkv, bc_qkv, wp_qkv, bp_qkv,
                                                 qkvc, qkvp);
  }

  // Kernel 2: fused FiLM MLPs + edge output + score sums
  k_main<<<B_ * N_ * 8, 256, 0, stream>>>(
      edges, qkvc, qkvp,
      fcmulT, fcaddT, fcm1T, fcm2T,
      fpmulT, fpaddT, fpm1T, fpm2T, weT,
      fc_mul_b, fc_add_b, fc_m1_b, fc_m2_b,
      fp_mul_b, fp_add_b, fp_m1_b, fp_m2_b,
      be_out, scoreC, scoreP, out_e);

  // Kernel 3: softmax + weighted V + out_c/out_p
  k_attn_out<<<B_ * N_, 256, 0, stream>>>(scoreC, scoreP, qkvc, qkvp,
                                          wc_out, bc_out, wp_out, bp_out,
                                          out_c, out_p);
}
